// combinelayer_39977555591552
// MI455X (gfx1250) — matrix operand order repaired and checked by reading, not compiled, not run on hardware
//
#include <hip/hip_runtime.h>
#include <hip/hip_bf16.h>
#include <cmath>

// ---------------------------------------------------------------------------
// GCNII-style layer, algebraically refactored:
//   T1f = diag(f) * (support1 @ input)          [4096,128]
//   T3  = adj @ input                           [4096,128]
//   SUP = (1-a)*(g*(support0 @ T1f) + (1-g)*T3) + a*h0
//   out = beta*(SUP @ weight) + (1-beta)*SUP + input,  beta = log(lamda/l + 1)
// Avoids the [4096,4096]x[4096,4096] GEMM (10x FLOP cut); op becomes HBM-bound
// (192 MB fp32 operand traffic @ 23.3 TB/s ~ 8.3 us floor).
// Compute: v_wmma_f32_16x16x32_bf16 (fp32->bf16 operands, fp32 accumulate).
// Data movement: TENSOR_LOAD_TO_LDS (TDM) with descriptor-driven LDS padding,
// double-buffered so the DMA of tile k+1 overlaps WMMAs of tile k.
// ---------------------------------------------------------------------------

typedef __attribute__((ext_vector_type(16))) __bf16 v16bf;
typedef __attribute__((ext_vector_type(8)))  float  v8f;
typedef __attribute__((ext_vector_type(4)))  unsigned int u32x4;
typedef __attribute__((ext_vector_type(8)))  int          i32x8;
typedef __attribute__((ext_vector_type(4)))  int          i32x4;

#define NN   4096
#define DCOL 128

#if defined(__has_builtin)
#  if __has_builtin(__builtin_amdgcn_tensor_load_to_lds)
#    define GCN_USE_TDM 1
#  endif
#endif

#ifdef GCN_USE_TDM
// Issue one 2-D TDM tile load: rows x rowLen fp32 elements, row stride
// tensorRowLen, with LDS padding of (padAmountEnc+1) dwords every
// 2^(padIntervalEnc+1) dwords (== one tile row).  D# layout per ISA §8.3-8.4.
// Groups 2/3 (and the trailing i32x8 operand of the 6-arg clang-23 builtin)
// are only needed for >=3-D tensors / gather mode -> zero-filled here.
__device__ __forceinline__ void tdm_load_2d(unsigned int lds_byte_addr,
                                            const float* gptr,
                                            int rowLen, int rows,
                                            int tensorRowLen, int tensorRows,
                                            unsigned int padIntervalEnc,
                                            unsigned int padAmountEnc)
{
    const unsigned long long ga = (unsigned long long)(size_t)gptr;
    u32x4 g0 = {};
    g0[0] = 1u;                                   // count=1, user descriptor
    g0[1] = lds_byte_addr;                        // lds_addr
    g0[2] = (unsigned int)ga;                     // global_addr[31:0]
    g0[3] = (unsigned int)((ga >> 32) & 0x01FFFFFFu) | (2u << 30); // addr[56:32] | type=2

    i32x8 g1 = {};
    // data_size=2 (4B) | pad_enable | pad_interval | pad_amount
    g1[0] = (int)((2u << 16) | (1u << 20) | (padIntervalEnc << 22) | (padAmountEnc << 25));
    g1[1] = (int)(((unsigned)tensorRowLen & 0xFFFFu) << 16);                 // tensor_dim0 lo16
    g1[2] = (int)((((unsigned)tensorRowLen >> 16) & 0xFFFFu) |
                  (((unsigned)tensorRows & 0xFFFFu) << 16));                 // dim0 hi | dim1 lo
    g1[3] = (int)((((unsigned)tensorRows >> 16) & 0xFFFFu) |
                  ((unsigned)rowLen << 16));                                 // dim1 hi | tile_dim0
    g1[4] = rows;                                  // tile_dim1 (tile_dim2 = 0)
    g1[5] = tensorRowLen;                          // tensor_dim0_stride lo32
    g1[6] = 0;
    g1[7] = 0;

    i32x4 z4 = {};
    i32x8 z8 = {};
    __builtin_amdgcn_tensor_load_to_lds(g0, g1, z4, z4, z8, 0);
}
#endif

// MODE 0: C = A@B, row-scaled by f        (T1f = diag(f)*(S1@X))
// MODE 1: C = A@B                         (T3 = adj@X)
// MODE 2: fused support epilogue on A@B   (SUP)
// MODE 3: beta*(A@B) + (1-beta)*aux1 + aux2  (final output)
template <int MODE>
__global__ __launch_bounds__(256)
void gcn_gemm(const float* __restrict__ A,     // [4096, K] row-major
              const float* __restrict__ B,     // [K, 128] row-major
              float* __restrict__ C,           // [4096, 128]
              int K,
              const float* __restrict__ fvec,
              const float* __restrict__ aux1,  // MODE2: T3   MODE3: SUP
              const float* __restrict__ aux2,  // MODE2: h0   MODE3: input
              const float* __restrict__ gamma_p,
              const float* __restrict__ alpha_p,
              const float* __restrict__ lamda_p,
              const int*   __restrict__ l_p)
{
    // Padded LDS strides -> bank-conflict-free WMMA fragment gathers
    // (64 banks, wave32): A rows 33 dwords apart, B rows 130 dwords apart.
    constexpr int AS = 33;           // 32 + 1 pad
    constexpr int BS = 130;          // 128 + 2 pad
    constexpr int ASZ = 64 * AS;     // one A buffer (fp32 elems)
    constexpr int BSZ = 32 * BS;     // one B buffer
    __shared__ float As[2 * ASZ];    // double buffered
    __shared__ float Bs[2 * BSZ];

    const int tid   = threadIdx.x;
    const int wave  = tid >> 5;          // 8 waves
    const int lane  = tid & 31;
    const int hk    = lane >> 4;         // K-half select (WMMA lane layout)
    const int l15   = lane & 15;
    const int mrow  = (wave & 3) * 16;   // 16-row strip inside 64-row tile
    const int nhalf = (wave >> 2) * 64;  // 64-column half of N=128
    const int rowBase = blockIdx.x * 64;
    const int nk = K / 32;

    v8f acc[4] = {};

#ifdef GCN_USE_TDM
    // LDS aperture maps generic addr[31:0] directly to the LDS address.
    const unsigned int asAddr = (unsigned int)(size_t)(void*)&As[0];
    const unsigned int bsAddr = (unsigned int)(size_t)(void*)&Bs[0];
    auto issue = [&](int step) {
        const int buf = step & 1;
        const int k0  = step * 32;
        // A tile: 64 rows x 32 dwords, pad 1 dword / 32 dwords  (enc 4 / 0)
        tdm_load_2d(asAddr + (unsigned)buf * ASZ * 4u,
                    &A[(size_t)rowBase * (size_t)K + (size_t)k0],
                    32, 64, K, NN, 4u, 0u);
        // B tile: 32 rows x 128 dwords, pad 2 dwords / 128 dwords (enc 6 / 1)
        tdm_load_2d(bsAddr + (unsigned)buf * BSZ * 4u,
                    &B[(size_t)k0 * (size_t)DCOL],
                    DCOL, 32, DCOL, K, 6u, 1u);
    };
    if (wave == 0) issue(0);
#endif

    for (int step = 0; step < nk; ++step) {
        const int buf = step & 1;
        const int k0  = step * 32;
        float* Asb = &As[buf * ASZ];
        float* Bsb = &Bs[buf * BSZ];

#ifdef GCN_USE_TDM
        if (wave == 0) {
            if (step + 1 < nk) {
                issue(step + 1);                       // prefetch next tile
                __builtin_amdgcn_s_wait_tensorcnt(2);  // current pair done (in-order)
            } else {
                __builtin_amdgcn_s_wait_tensorcnt(0);
            }
        }
        __syncthreads();   // release all waves: LDS tile `buf` is valid
#else
        // Fallback: cooperative global->LDS staging
#pragma unroll
        for (int i = 0; i < 2; ++i) {          // A tile: 64x32 fp32
            const int q = tid + i * 256;
            const int r = q >> 3;
            const int c = (q & 7) * 4;
            const float4 v = *reinterpret_cast<const float4*>(
                &A[(size_t)(rowBase + r) * (size_t)K + (size_t)(k0 + c)]);
            float* p = &Asb[r * AS + c];
            p[0] = v.x; p[1] = v.y; p[2] = v.z; p[3] = v.w;
        }
#pragma unroll
        for (int i = 0; i < 4; ++i) {          // B tile: 32x128 fp32
            const int q = tid + i * 256;
            const int r = q >> 5;
            const int c = (q & 31) * 4;
            const float4 v = *reinterpret_cast<const float4*>(
                &B[(size_t)(k0 + r) * (size_t)DCOL + (size_t)c]);
            float* p = &Bsb[r * BS + c];
            p[0] = v.x; p[1] = v.y; p[2] = v.z; p[3] = v.w;
        }
        __syncthreads();
#endif

        // ---- A fragment: 16x32 bf16, contiguous k = 16*hk + j order, same as B ----
        v16bf afrag;
        {
            const float* ap = &Asb[(mrow + l15) * AS + hk * 16];
#pragma unroll
            for (int j = 0; j < 8; ++j) afrag[j]     = (__bf16)ap[j];
#pragma unroll
            for (int j = 0; j < 8; ++j) afrag[8 + j] = (__bf16)ap[8 + j];
        }

#pragma unroll
        for (int nt = 0; nt < 4; ++nt) {
            v16bf bfrag;
            const float* bp = &Bsb[(hk * 16) * BS + nhalf + nt * 16 + l15];
#pragma unroll
            for (int j = 0; j < 16; ++j) bfrag[j] = (__bf16)bp[j * BS];

            acc[nt] = __builtin_amdgcn_wmma_f32_16x16x32_bf16(
                false, afrag, false, bfrag, (short)0, acc[nt], false, false);
        }
        __syncthreads();   // all waves done with `buf` before it is refilled
    }

    // ---- epilogue ----
    float gamma = 0.f, alpha = 0.f, beta = 0.f;
    if constexpr (MODE == 2) { gamma = gamma_p[0]; alpha = alpha_p[0]; }
    if constexpr (MODE == 3) { beta = logf(lamda_p[0] / (float)l_p[0] + 1.0f); }

#pragma unroll
    for (int nt = 0; nt < 4; ++nt) {
#pragma unroll
        for (int r = 0; r < 8; ++r) {
            // C/D layout: VGPR r holds M=r (lanes 0-15) and M=r+8 (lanes 16-31)
            const int row = rowBase + mrow + hk * 8 + r;
            const int col = nhalf + nt * 16 + l15;
            const size_t idx = (size_t)row * DCOL + col;
            float v = acc[nt][r];
            if constexpr (MODE == 0) {
                v *= fvec[row];                       // fold diag(f) into T1
            } else if constexpr (MODE == 2) {
                v = (1.f - alpha) * (gamma * v + (1.f - gamma) * aux1[idx])
                    + alpha * aux2[idx];
            } else if constexpr (MODE == 3) {
                v = beta * v + (1.f - beta) * aux1[idx] + aux2[idx];
            }
            C[idx] = v;
        }
    }
}

extern "C" void kernel_launch(void* const* d_in, const int* in_sizes, int n_in,
                              void* d_out, int out_size, void* d_ws, size_t ws_size,
                              hipStream_t stream) {
    const float* input    = (const float*)d_in[0];
    const float* support0 = (const float*)d_in[1];
    const float* support1 = (const float*)d_in[2];
    const float* adj      = (const float*)d_in[3];
    const float* h0       = (const float*)d_in[4];
    const float* fvec     = (const float*)d_in[5];
    const float* weight   = (const float*)d_in[6];
    const float* gamma_p  = (const float*)d_in[7];
    const float* lamda_p  = (const float*)d_in[8];
    const float* alpha_p  = (const float*)d_in[9];
    const int*   l_p      = (const int*)d_in[10];

    // workspace: 3 x [4096,128] fp32 = 6 MB
    float* T1f = (float*)d_ws;
    float* T3  = T1f + (size_t)NN * DCOL;
    float* SUP = T3  + (size_t)NN * DCOL;
    float* out = (float*)d_out;

    const dim3 grid(NN / 64), block(256);

    // T1f = diag(f) * (support1 @ input)
    gcn_gemm<0><<<grid, block, 0, stream>>>(support1, input, T1f, NN,
        fvec, nullptr, nullptr, gamma_p, alpha_p, lamda_p, l_p);
    // T3 = adj @ input
    gcn_gemm<1><<<grid, block, 0, stream>>>(adj, input, T3, NN,
        fvec, nullptr, nullptr, gamma_p, alpha_p, lamda_p, l_p);
    // SUP = (1-a)*(g*(support0@T1f) + (1-g)*T3) + a*h0
    gcn_gemm<2><<<grid, block, 0, stream>>>(support0, T1f, SUP, NN,
        fvec, T3, h0, gamma_p, alpha_p, lamda_p, l_p);
    // out = beta*(SUP@weight) + (1-beta)*SUP + input
    gcn_gemm<3><<<grid, block, 0, stream>>>(SUP, weight, out, DCOL,
        fvec, SUP, input, gamma_p, alpha_p, lamda_p, l_p);
}
